// EGNN_360777253437
// MI455X (gfx1250) — matrix operand order repaired and checked by reading, not compiled, not run on hardware
//
#include <hip/hip_runtime.h>
#include <cmath>

typedef __attribute__((ext_vector_type(16))) __bf16 v16bf;
typedef __attribute__((ext_vector_type(8)))  float  v8f;

#define DIM   128
#define MD    16
#define KNNK  32
#define NBAT  2
#define NPTS  4096
#define NNODE (NBAT*NPTS)
#define NEDGE (NNODE*KNNK)

// edge GEMM1: K 265->288 (9 kt), N 530->576 (36 nt, fully uniform across 4 N-groups)
#define KT1  9
#define NT1P 36
#define K1P  288
#define N1PP 576
// edge GEMM2: K 576 (18 kt), N 16
#define KT2  18
// node GEMM1: K 144->160 (5 kt), N 256 (16 nt)
#define KTn1 5
#define NTn1 16
#define KnP  160
// node GEMM2: K 256 (8 kt), N 128 (8 nt)
#define KTn2 8
#define NTn2 8

// edge kernel blocking: 4 nodes = 128 edges = 8 M-tiles per 1024-thread block
#define NPB   4
#define ROWS  128

// ---- workspace layout (bytes) ----
#define W1F_OFF    0
#define W1F_TILES  (KT1*NT1P)                        // 324
#define W2F_OFF    (W1F_OFF + W1F_TILES*512*2)       // 331776
#define W2F_TILES  (KT2)                             // 18
#define WN1F_OFF   (W2F_OFF + W2F_TILES*512*2)       // 350208
#define WN1F_TILES (KTn1*NTn1)                       // 80
#define WN2F_OFF   (WN1F_OFF + WN1F_TILES*512*2)     // 432128
#define WN2F_TILES (KTn2*NTn2)                       // 64
#define TILES_TOT  (W1F_TILES+W2F_TILES+WN1F_TILES+WN2F_TILES) // 486
#define IDX_OFF    (WN2F_OFF + WN2F_TILES*512*2)     // 497664
#define RELD_OFF   (IDX_OFF + NEDGE*4)               // rel x,y,z,d per edge (16B)
#define MI_OFF     (RELD_OFF + NEDGE*16)

__device__ __forceinline__ float silu_f(float x) { return x / (1.0f + __expf(-x)); }

__device__ __forceinline__ unsigned long long shfl_xor_u64(unsigned long long v, int mask) {
    unsigned lo = (unsigned)v, hi = (unsigned)(v >> 32);
    lo = __shfl_xor(lo, mask, 32);
    hi = __shfl_xor(hi, mask, 32);
    return ((unsigned long long)hi << 32) | lo;
}

// ---------------- kernel 0: repack weights into bf16 WMMA B-fragment layout ----------------
// B fragment (16-bit, 32x16 KxN): lane = N (lane&15), K half by lane>=16, 16 K elems/lane.
// Lane-major per tile: dst[(tile*32 + lane)*16 + e], tile = kt*NT + nt. Zero-padded.
__global__ void __launch_bounds__(256) pack_weights_kernel(
    const float* __restrict__ We1, const float* __restrict__ We2,
    const float* __restrict__ Wn1, const float* __restrict__ Wn2, char* wsc)
{
    int t = blockIdx.x * 256 + threadIdx.x;
    int tile = t >> 5, lane = t & 31;
    if (tile >= TILES_TOT) return;
    int kh = (lane >= 16) ? 8 : 0;
    int nl = lane & 15;
    __bf16* dst; const float* W; int kt, nt, Kd, Nd, ldw;
    if (tile < W1F_TILES) {
        kt = tile / NT1P; nt = tile % NT1P;
        dst = (__bf16*)(wsc + W1F_OFF) + (size_t)tile * 512;
        W = We1; Kd = 265; Nd = 530; ldw = 530;
    } else if (tile < W1F_TILES + W2F_TILES) {
        int tt = tile - W1F_TILES; kt = tt; nt = 0;
        dst = (__bf16*)(wsc + W2F_OFF) + (size_t)tt * 512;
        W = We2; Kd = 530; Nd = 16; ldw = 16;
    } else if (tile < W1F_TILES + W2F_TILES + WN1F_TILES) {
        int tt = tile - (W1F_TILES + W2F_TILES);
        kt = tt / NTn1; nt = tt % NTn1;
        dst = (__bf16*)(wsc + WN1F_OFF) + (size_t)tt * 512;
        W = Wn1; Kd = 144; Nd = 256; ldw = 256;
    } else {
        int tt = tile - (W1F_TILES + W2F_TILES + WN1F_TILES);
        kt = tt / NTn2; nt = tt % NTn2;
        dst = (__bf16*)(wsc + WN2F_OFF) + (size_t)tt * 512;
        W = Wn2; Kd = 256; Nd = 128; ldw = 128;
    }
    int n = nt * 16 + nl;
    #pragma unroll
    for (int e = 0; e < 16; ++e) {
        int k = kt * 32 + kh + ((e < 8) ? e : e + 8);
        float v = (k < Kd && n < Nd) ? W[(size_t)k * ldw + n] : 0.0f;
        dst[lane * 16 + e] = (__bf16)v;
    }
}

// ---------------- kernel 1: KNN (one block per node, cached-local-best selection) ----------------
__global__ void __launch_bounds__(128) knn_kernel(
    const float* __restrict__ coors, int* __restrict__ idxb, float* __restrict__ reld)
{
    __shared__ float dist[NPTS];
    __shared__ unsigned long long lred[4];
    __shared__ unsigned long long sWin;
    int g = blockIdx.x;
    int i = g & (NPTS - 1);
    int base = (g >> 12) * NPTS;
    int tid = threadIdx.x;
    float cx = coors[(size_t)(base + i) * 3 + 0];
    float cy = coors[(size_t)(base + i) * 3 + 1];
    float cz = coors[(size_t)(base + i) * 3 + 2];
    for (int s = 0; s < NPTS / 128; ++s) {
        int j = tid + s * 128;
        float dx = cx - coors[(size_t)(base + j) * 3 + 0];
        float dy = cy - coors[(size_t)(base + j) * 3 + 1];
        float dz = cz - coors[(size_t)(base + j) * 3 + 2];
        dist[j] = dx * dx + dy * dy + dz * dz;
    }
    __syncthreads();
    // per-thread cached best over its stripe (dist-bits<<32 | idx: min => nearest, lowest idx)
    unsigned long long best = ~0ull;
    for (int s = 0; s < NPTS / 128; ++s) {
        int j = tid + s * 128;
        unsigned long long key = ((unsigned long long)__float_as_uint(dist[j]) << 32) | (unsigned)j;
        if (key < best) best = key;
    }
    int lane = tid & 31, w = tid >> 5;
    for (int t = 0; t < KNNK; ++t) {
        unsigned long long r = best;
        for (int m = 16; m >= 1; m >>= 1) {
            unsigned long long o = shfl_xor_u64(r, m);
            if (o < r) r = o;
        }
        if (lane == 0) lred[w] = r;
        __syncthreads();
        if (tid == 0) {
            unsigned long long bb = lred[0];
            for (int q = 1; q < 4; ++q) if (lred[q] < bb) bb = lred[q];
            sWin = bb;
            int j = (int)(bb & 0xffffffffu);
            float d = __uint_as_float((unsigned)(bb >> 32));
            idxb[(size_t)g * KNNK + t] = j;
            float* rd = reld + ((size_t)g * KNNK + t) * 4;
            rd[0] = cx - coors[(size_t)(base + j) * 3 + 0];
            rd[1] = cy - coors[(size_t)(base + j) * 3 + 1];
            rd[2] = cz - coors[(size_t)(base + j) * 3 + 2];
            rd[3] = d;
        }
        __syncthreads();
        int jw = (int)(sWin & 0xffffffffu);
        if ((jw & 127) == tid) {       // only the owner rescans its stripe
            dist[jw] = 3.4e38f;
            best = ~0ull;
            for (int s = 0; s < NPTS / 128; ++s) {
                int j = tid + s * 128;
                unsigned long long key = ((unsigned long long)__float_as_uint(dist[j]) << 32) | (unsigned)j;
                if (key < best) best = key;
            }
        }
    }
}

// Load a 16-bit A fragment (16x32) from row-major LDS (even stride) per ISA layout:
// lane: M = lane&15, K half (+8) for lanes 16-31; VGPR v: K = 2(v&3)+(v>=4?16:0), pairs contiguous.
__device__ __forceinline__ v16bf load_afrag(const __bf16* src, int stride, int row, int ktbase, int lane) {
    union { v16bf v; unsigned u[8]; } af;
    int khalf = (lane >= 16) ? 8 : 0;
    #pragma unroll
    for (int vv = 0; vv < 8; ++vv) {
        int k0 = ktbase + khalf + ((vv >= 4) ? 16 : 0) + 2 * (vv & 3);
        af.u[vv] = *reinterpret_cast<const unsigned*>(&src[row * stride + k0]);
    }
    return af.v;
}

// ---------------- kernel 2: edge pipeline (one block = 4 nodes = 128 edges = 8 M-tiles) ----------------
__global__ void __launch_bounds__(1024) edge_kernel(
    const float* __restrict__ feats, const float* __restrict__ coors,
    const float* __restrict__ be1, const float* __restrict__ be2,
    const float* __restrict__ Wc1, const float* __restrict__ bc1,
    const float* __restrict__ Wc2, const float* __restrict__ bc2,
    const char* __restrict__ wsc, float* __restrict__ m_i, float* __restrict__ out_coors)
{
    __shared__ __align__(16) __bf16 sA[ROWS * K1P];    // 73728 B
    __shared__ __align__(16) __bf16 sH[ROWS * N1PP];   // 147456 B
    __shared__ float sM[ROWS * 16];                    // 8192 B
    __shared__ float sRel[ROWS * 4];                   // 2048 B
    __shared__ float sRed[ROWS * 8];                   // 4096 B
    __shared__ float sW[ROWS];                         // 512 B
    __shared__ int   sJ[ROWS];                         // 512 B

    const int g0 = blockIdx.x * NPB;                   // first node of block
    const int tid = threadIdx.x;
    const int* idxb = (const int*)(wsc + IDX_OFF);
    const float* reld = (const float*)(wsc + RELD_OFF);

    if (tid < ROWS * 4) sRel[tid] = reld[(size_t)g0 * KNNK * 4 + tid];
    if (tid < ROWS)     sJ[tid]   = idxb[(size_t)g0 * KNNK + tid];
    __syncthreads();

    // ---- build 128x288 bf16 edge_in tile ----
    for (int x = tid; x < ROWS * K1P; x += 1024) {
        int r = x / K1P, c = x % K1P;
        int gnode = g0 + (r >> 5);
        int i = gnode & (NPTS - 1);
        int nbase = (gnode >> 12) * NPTS;
        float v = 0.0f;
        if (c < 128) {
            v = feats[(size_t)(nbase + i) * DIM + c];
        } else if (c < 256) {
            v = feats[(size_t)(nbase + sJ[r]) * DIM + (c - 128)];
        } else if (c < 265) {
            int s = c - 256;
            float d = sRel[r * 4 + 3];
            if (s < 4)      v = __sinf(d / (float)(1 << s));
            else if (s < 8) v = __cosf(d / (float)(1 << (s - 4)));
            else            v = d;
        }
        sA[x] = (__bf16)v;
    }
    __syncthreads();

    const int lane = tid & 31, w = tid >> 5;           // 32 waves
    const int mt = w & 7, ng = w >> 3;                 // 8 M-tiles x 4 N-groups (9 tiles each, uniform)
    const v8f vzero = {0.f,0.f,0.f,0.f,0.f,0.f,0.f,0.f};

    // ---- GEMM1: (128x288) x (288x576) ----
    const v16bf* W1f = reinterpret_cast<const v16bf*>(wsc + W1F_OFF);
    v8f acc[9];
    #pragma unroll
    for (int q = 0; q < 9; ++q) acc[q] = vzero;
    const int row = (lane & 15) + mt * 16;
    for (int kt = 0; kt < KT1; ++kt) {
        v16bf a = load_afrag(sA, K1P, row, kt * 32, lane);
        const v16bf* bp = W1f + ((size_t)(kt * NT1P + ng) * 32 + lane);
        #pragma unroll
        for (int q = 0; q < 9; ++q) {                  // nt = ng + 4q, constant offsets
            v16bf bf = bp[q * 4 * 32];
            acc[q] = __builtin_amdgcn_wmma_f32_16x16x32_bf16(
                false, a, false, bf, (short)0, acc[q], false, false);
        }
    }
    #pragma unroll
    for (int q = 0; q < 9; ++q) {
        int ncol = (ng + 4 * q) * 16 + (lane & 15);
        float bias = (ncol < 530) ? be1[ncol] : 0.0f;
        #pragma unroll
        for (int vv = 0; vv < 8; ++vv) {
            int m = vv + ((lane >= 16) ? 8 : 0) + mt * 16;
            sH[m * N1PP + ncol] = (__bf16)silu_f(acc[q][vv] + bias);
        }
    }
    __syncthreads();

    // ---- GEMM2: (128x576) x (576x16) -> m_ij (waves 0..7, one M-tile each) ----
    if (w < 8) {
        const v16bf* W2f = reinterpret_cast<const v16bf*>(wsc + W2F_OFF);
        v8f acc2 = vzero;
        int row2 = (lane & 15) + w * 16;
        for (int kt = 0; kt < KT2; ++kt) {
            v16bf a = load_afrag(sH, N1PP, row2, kt * 32, lane);
            v16bf bf = W2f[kt * 32 + lane];
            acc2 = __builtin_amdgcn_wmma_f32_16x16x32_bf16(
                false, a, false, bf, (short)0, acc2, false, false);
        }
        int n = lane & 15;
        float bias = be2[n];
        #pragma unroll
        for (int vv = 0; vv < 8; ++vv) {
            int m = vv + ((lane >= 16) ? 8 : 0) + w * 16;
            sM[m * 16 + n] = silu_f(acc2[vv] + bias);
        }
    }
    __syncthreads();

    // ---- coors MLP: 16->64 SiLU ->1 (8 threads per edge, 128 edges) ----
    {
        int e = tid >> 3, u8 = tid & 7;
        float partial = 0.0f;
        #pragma unroll
        for (int uu = 0; uu < 8; ++uu) {
            int u = u8 * 8 + uu;
            float a = bc1[u];
            #pragma unroll
            for (int k2 = 0; k2 < 16; ++k2) a += sM[e * 16 + k2] * Wc1[k2 * 64 + u];
            partial += silu_f(a) * Wc2[u];
        }
        sRed[e * 8 + u8] = partial;
    }
    __syncthreads();
    if (tid < ROWS) {
        float wsum = bc2[0];
        #pragma unroll
        for (int q = 0; q < 8; ++q) wsum += sRed[tid * 8 + q];
        sW[tid] = wsum;
    }
    __syncthreads();
    if (tid < NPB * 16) {   // m_i = sum_k m_ij per node
        int nloc = tid >> 4, n = tid & 15;
        float s = 0.0f;
        for (int e = 0; e < 32; ++e) s += sM[(nloc * 32 + e) * 16 + n];
        m_i[(size_t)(g0 + nloc) * 16 + n] = s;
    }
    if (tid < NPB * 3) {    // coors_out = sum_k w*rel + coors
        int nloc = tid / 3, c = tid % 3;
        int gnode = g0 + nloc;
        int i = gnode & (NPTS - 1);
        int nbase = (gnode >> 12) * NPTS;
        float s = coors[(size_t)(nbase + i) * 3 + c];
        for (int e = 0; e < 32; ++e) s += sW[nloc * 32 + e] * sRel[(nloc * 32 + e) * 4 + c];
        out_coors[(size_t)gnode * 3 + c] = s;
    }
}

// ---------------- kernel 3: node MLP (16 nodes per block) ----------------
__global__ void __launch_bounds__(256) node_kernel(
    const float* __restrict__ feats, const float* __restrict__ bn1,
    const float* __restrict__ bn2, const char* __restrict__ wsc,
    const float* __restrict__ m_i, float* __restrict__ out_node)
{
    __shared__ __align__(16) __bf16 sA[16 * KnP];   // 5120 B
    __shared__ __align__(16) __bf16 sH[16 * 256];   // 8192 B
    const int nb = blockIdx.x * 16;
    const int tid = threadIdx.x;

    for (int x = tid; x < 16 * KnP; x += 256) {
        int r = x / KnP, c = x % KnP;
        int node = nb + r;
        float v = 0.0f;
        if (c < 128)      v = feats[(size_t)node * DIM + c];
        else if (c < 144) v = m_i[(size_t)node * 16 + (c - 128)];
        sA[x] = (__bf16)v;
    }
    __syncthreads();

    const int lane = tid & 31, w = tid >> 5;
    const v8f vzero = {0.f,0.f,0.f,0.f,0.f,0.f,0.f,0.f};
    const int row = lane & 15;

    // GEMM1: (16x160) x (160x256), 2 N-tiles per wave
    const v16bf* Wn1f = reinterpret_cast<const v16bf*>(wsc + WN1F_OFF);
    v8f acc[2]; acc[0] = vzero; acc[1] = vzero;
    for (int kt = 0; kt < KTn1; ++kt) {
        v16bf a = load_afrag(sA, KnP, row, kt * 32, lane);
        #pragma unroll
        for (int q = 0; q < 2; ++q) {
            int nt = w * 2 + q;
            v16bf bf = Wn1f[(kt * NTn1 + nt) * 32 + lane];
            acc[q] = __builtin_amdgcn_wmma_f32_16x16x32_bf16(
                false, a, false, bf, (short)0, acc[q], false, false);
        }
    }
    #pragma unroll
    for (int q = 0; q < 2; ++q) {
        int ncol = (w * 2 + q) * 16 + (lane & 15);
        float bias = bn1[ncol];
        #pragma unroll
        for (int vv = 0; vv < 8; ++vv) {
            int m = vv + ((lane >= 16) ? 8 : 0);
            sH[m * 256 + ncol] = (__bf16)silu_f(acc[q][vv] + bias);
        }
    }
    __syncthreads();

    // GEMM2: (16x256) x (256x128) + bias + residual
    const v16bf* Wn2f = reinterpret_cast<const v16bf*>(wsc + WN2F_OFF);
    v8f acc2 = vzero;
    for (int kt = 0; kt < KTn2; ++kt) {
        v16bf a = load_afrag(sH, 256, row, kt * 32, lane);
        v16bf bf = Wn2f[(kt * NTn2 + w) * 32 + lane];
        acc2 = __builtin_amdgcn_wmma_f32_16x16x32_bf16(
            false, a, false, bf, (short)0, acc2, false, false);
    }
    {
        int ncol = w * 16 + (lane & 15);
        float bias = bn2[ncol];
        #pragma unroll
        for (int vv = 0; vv < 8; ++vv) {
            int m = vv + ((lane >= 16) ? 8 : 0);
            int node = nb + m;
            out_node[(size_t)node * DIM + ncol] =
                acc2[vv] + bias + feats[(size_t)node * DIM + ncol];
        }
    }
}

extern "C" void kernel_launch(void* const* d_in, const int* in_sizes, int n_in,
                              void* d_out, int out_size, void* d_ws, size_t ws_size,
                              hipStream_t stream)
{
    const float* feats = (const float*)d_in[0];
    const float* coors = (const float*)d_in[1];
    const float* We1   = (const float*)d_in[2];
    const float* be1   = (const float*)d_in[3];
    const float* We2   = (const float*)d_in[4];
    const float* be2   = (const float*)d_in[5];
    const float* Wc1   = (const float*)d_in[6];
    const float* bc1   = (const float*)d_in[7];
    const float* Wc2   = (const float*)d_in[8];
    const float* bc2   = (const float*)d_in[9];
    const float* Wn1   = (const float*)d_in[10];
    const float* bn1   = (const float*)d_in[11];
    const float* Wn2   = (const float*)d_in[12];
    const float* bn2   = (const float*)d_in[13];
    char* wsc = (char*)d_ws;
    float* out_node  = (float*)d_out;
    float* out_coors = out_node + (size_t)NNODE * DIM;

    pack_weights_kernel<<<(TILES_TOT * 32 + 255) / 256, 256, 0, stream>>>(We1, We2, Wn1, Wn2, wsc);
    knn_kernel<<<NNODE, 128, 0, stream>>>(coors, (int*)(wsc + IDX_OFF), (float*)(wsc + RELD_OFF));
    edge_kernel<<<NNODE / NPB, 1024, 0, stream>>>(feats, coors, be1, be2, Wc1, bc1, Wc2, bc2,
                                                  wsc, (float*)(wsc + MI_OFF), out_coors);
    node_kernel<<<NNODE / 16, 256, 0, stream>>>(feats, bn1, bn2, wsc, (float*)(wsc + MI_OFF), out_node);
}
